// Attention_46222438039802
// MI455X (gfx1250) — matrix operand order repaired and checked by reading, not compiled, not run on hardware
//
#include <hip/hip_runtime.h>
#include <hip/hip_bf16.h>
#include <stdint.h>

typedef __bf16 bf16_t;
typedef bf16_t v16bf __attribute__((ext_vector_type(16)));
typedef bf16_t v8bf  __attribute__((ext_vector_type(8)));
typedef bf16_t v4bf  __attribute__((ext_vector_type(4)));
typedef float  v8f   __attribute__((ext_vector_type(8)));
typedef float  v4f   __attribute__((ext_vector_type(4)));
typedef unsigned int v4u __attribute__((ext_vector_type(4)));
typedef int    v8i   __attribute__((ext_vector_type(8)));
typedef int    v4i   __attribute__((ext_vector_type(4)));

#define F_DIM 1024
#define C_DIM 64
#define M_TOTAL 16384           // B*S = 8*2048
#define LDS_STRIDE 80           // bf16 elems per row: 160B, 16B-aligned rows

#if __has_builtin(__builtin_amdgcn_tensor_load_to_lds)
#define HAVE_TDM 1
#else
#define HAVE_TDM 0
#endif

union Frag { v16bf v; v8bf h[2]; };

// ---------------- prep: f32 -> bf16 elementwise (inputs) ----------------
__global__ void k_convert_bf16(const float* __restrict__ in, bf16_t* __restrict__ out, int n4) {
    int i = blockIdx.x * blockDim.x + threadIdx.x;
    if (i < n4) {
        v4f v = ((const v4f*)in)[i];
        v4bf o;
        o[0] = (bf16_t)v[0]; o[1] = (bf16_t)v[1];
        o[2] = (bf16_t)v[2]; o[3] = (bf16_t)v[3];
        ((v4bf*)out)[i] = o;
    }
}

// ---------------- prep: WQ (K x N f32) -> WQ^T (N x K bf16) ----------------
__global__ void k_transpose_wq(const float* __restrict__ wq, bf16_t* __restrict__ wqt) {
    __shared__ float tile[32][33];
    int x = threadIdx.x, y = threadIdx.y;           // block (32,8)
    int bx = blockIdx.x * 32, by = blockIdx.y * 32;
    #pragma unroll
    for (int j = 0; j < 32; j += 8)
        tile[y + j][x] = wq[(size_t)(by + y + j) * F_DIM + bx + x];
    __syncthreads();
    #pragma unroll
    for (int j = 0; j < 32; j += 8)
        wqt[(size_t)(bx + y + j) * F_DIM + by + x] = (bf16_t)tile[x][y + j];
}

// ------- prep: K = label_emb@WK (C x F, bf16), V^T (F x C, bf16) -------
__global__ void k_prep_kv(const float* __restrict__ lab, const float* __restrict__ wk,
                          const float* __restrict__ wv, bf16_t* __restrict__ kbf,
                          bf16_t* __restrict__ vtb) {
    __shared__ float lrow[F_DIM];
    int c = blockIdx.y;
    int g = blockIdx.x * 256 + threadIdx.x;
    for (int f = threadIdx.x; f < F_DIM; f += 256)
        lrow[f] = lab[(size_t)c * F_DIM + f];
    __syncthreads();
    float accK = 0.f, accV = 0.f;
    for (int f = 0; f < F_DIM; ++f) {
        float l = lrow[f];
        accK += l * wk[(size_t)f * F_DIM + g];
        accV += l * wv[(size_t)f * F_DIM + g];
    }
    kbf[(size_t)c * F_DIM + g] = (bf16_t)accK;   // row-major C x F == B^T for scores
    vtb[(size_t)g * C_DIM + c] = (bf16_t)accV;   // V^T, F x C == B^T for out GEMM
}

#if HAVE_TDM
// --- Tensor Data Mover: 2D tile (tile_d0 contiguous elems x tile_d1 rows) -> LDS ---
// D# per CDNA5 ISA ch.8: group0 = {flags, lds_addr, global_addr[56:0], type=2},
// group1 = {mask/data_size, tensor_dim0/1, tile_dim0/1/2, dim0_stride}, groups 2/3 zero (2-D).
__device__ __forceinline__ void tdm_load_2d(unsigned lds_off, const void* gptr,
                                            unsigned tensor_d0, unsigned tensor_d1,
                                            unsigned tile_d0, unsigned tile_d1,
                                            unsigned stride0) {
    unsigned long long ga = (unsigned long long)(uintptr_t)gptr;
    v4u g0;
    g0[0] = 1u;                                            // count=1 (valid), user mode
    g0[1] = lds_off;                                       // lds_addr (bytes)
    g0[2] = (unsigned)(ga & 0xffffffffu);                  // global_addr[31:0]
    g0[3] = (unsigned)((ga >> 32) & 0x01ffffffu) | (2u << 30); // addr[56:32], type=2
    v8i g1;
    g1[0] = (int)(1u << 16);                               // wg_mask=0, data_size=1 (2B)
    g1[1] = (int)((tensor_d0 & 0xffffu) << 16);            // tensor_dim0[15:0] @ [63:48]
    g1[2] = (int)(((tensor_d0 >> 16) & 0xffffu) | ((tensor_d1 & 0xffffu) << 16));
    g1[3] = (int)(((tensor_d1 >> 16) & 0xffffu) | ((tile_d0 & 0xffffu) << 16));
    g1[4] = (int)(tile_d1 & 0xffffu);                      // tile_dim1, tile_dim2=0
    g1[5] = (int)stride0;                                  // tensor_dim0_stride[31:0]
    g1[6] = 0;                                             // stride0[47:32], stride1[15:0]
    g1[7] = 0;
    v4i gz = {0, 0, 0, 0};
#if __has_include(<hip/amd_detail/amd_gfx1250_TDM.h>)
    v8i gz8 = {0, 0, 0, 0, 0, 0, 0, 0};
    __builtin_amdgcn_tensor_load_to_lds(g0, g1, gz, gz, gz8, 0);
#else
    __builtin_amdgcn_tensor_load_to_lds(g0, g1, gz, gz, 0);
#endif
}
#endif // HAVE_TDM

// ---------------- GEMM1: Qbf = Abf @ WQ^T  (M=16384,N=1024,K=1024) ----------------
// 128x128 block tile, 8 waves (2x4), wave tile 64x32 = 8 WMMA accumulators.
__global__ __launch_bounds__(256) void k_gemm_q(const bf16_t* __restrict__ A,
                                                const bf16_t* __restrict__ Bt,
                                                bf16_t* __restrict__ Q) {
    const int lane = threadIdx.x & 31;
    const int w    = threadIdx.x >> 5;      // 8 waves
    const int wm   = w >> 2;                // 0..1  (64-row stripe)
    const int wn   = w & 3;                 // 0..3  (32-col stripe)
    const int rowT = blockIdx.x * 128;
    const int colT = blockIdx.y * 128;
    const int lm = lane & 15;
    const int lh = lane >> 4;

    v8f acc[4][2];
    #pragma unroll
    for (int i = 0; i < 4; ++i)
        #pragma unroll
        for (int j = 0; j < 2; ++j)
            acc[i][j] = (v8f){0.f,0.f,0.f,0.f,0.f,0.f,0.f,0.f};

#if HAVE_TDM
    // Double-buffered 128x32 bf16 tiles staged by the Tensor Data Mover.
    __shared__ bf16_t lA[2][128 * 32];
    __shared__ bf16_t lB[2][128 * 32];
    unsigned offA[2], offB[2];
    offA[0] = (unsigned)(uintptr_t)&lA[0][0];
    offA[1] = (unsigned)(uintptr_t)&lA[1][0];
    offB[0] = (unsigned)(uintptr_t)&lB[0][0];
    offB[1] = (unsigned)(uintptr_t)&lB[1][0];

    if (w == 0) {   // one wave issues the DMA (TDM ignores EXEC; avoid duplicates)
        tdm_load_2d(offA[0], A  + (size_t)rowT * F_DIM, F_DIM, M_TOTAL, 32, 128, F_DIM);
        tdm_load_2d(offB[0], Bt + (size_t)colT * F_DIM, F_DIM, F_DIM,   32, 128, F_DIM);
        __builtin_amdgcn_s_wait_tensorcnt(0);
    }
    __syncthreads();

    for (int kt = 0; kt < F_DIM / 32; ++kt) {
        const int cur = kt & 1;
        const int nxt = cur ^ 1;
        if (w == 0 && kt + 1 < F_DIM / 32) {
            const int kk = (kt + 1) * 32;
            tdm_load_2d(offA[nxt], A  + (size_t)rowT * F_DIM + kk, F_DIM, M_TOTAL, 32, 128, F_DIM);
            tdm_load_2d(offB[nxt], Bt + (size_t)colT * F_DIM + kk, F_DIM, F_DIM,   32, 128, F_DIM);
        }
        // fragments from LDS (tiles are 128 rows x 32 k, row-major)
        Frag a[4], b[2];
        #pragma unroll
        for (int mi = 0; mi < 4; ++mi) {
            const bf16_t* p = &lA[cur][(wm * 64 + mi * 16 + lm) * 32 + lh * 8];
            a[mi].h[0] = *(const v8bf*)p;
            a[mi].h[1] = *(const v8bf*)(p + 16);
        }
        #pragma unroll
        for (int ni = 0; ni < 2; ++ni) {
            const bf16_t* p = &lB[cur][(wn * 32 + ni * 16 + lm) * 32 + lh * 8];
            b[ni].h[0] = *(const v8bf*)p;
            b[ni].h[1] = *(const v8bf*)(p + 16);
        }
        #pragma unroll
        for (int mi = 0; mi < 4; ++mi)
            #pragma unroll
            for (int ni = 0; ni < 2; ++ni)
                acc[mi][ni] = __builtin_amdgcn_wmma_f32_16x16x32_bf16(
                    false, a[mi].v, false, b[ni].v, (short)0, acc[mi][ni], false, false);
        if (w == 0 && kt + 1 < F_DIM / 32)
            __builtin_amdgcn_s_wait_tensorcnt(0);   // prefetch landed before releasing barrier
        __syncthreads();
    }
#else
    // Fallback: direct per-lane fragment loads from global (layouts per ISA 7.12.2).
    for (int kk = 0; kk < F_DIM; kk += 32) {
        Frag a[4], b[2];
        #pragma unroll
        for (int mi = 0; mi < 4; ++mi) {
            const bf16_t* p = A + (size_t)(rowT + wm * 64 + mi * 16 + lm) * F_DIM + kk + lh * 8;
            a[mi].h[0] = *(const v8bf*)p;
            a[mi].h[1] = *(const v8bf*)(p + 16);
        }
        #pragma unroll
        for (int ni = 0; ni < 2; ++ni) {
            const bf16_t* p = Bt + (size_t)(colT + wn * 32 + ni * 16 + lm) * F_DIM + kk + lh * 8;
            b[ni].h[0] = *(const v8bf*)p;
            b[ni].h[1] = *(const v8bf*)(p + 16);
        }
        #pragma unroll
        for (int mi = 0; mi < 4; ++mi)
            #pragma unroll
            for (int ni = 0; ni < 2; ++ni)
                acc[mi][ni] = __builtin_amdgcn_wmma_f32_16x16x32_bf16(
                    false, a[mi].v, false, b[ni].v, (short)0, acc[mi][ni], false, false);
    }
#endif

    // C layout: n = lane%16, m = (lane/16)*8 + vgpr
    #pragma unroll
    for (int mi = 0; mi < 4; ++mi)
        #pragma unroll
        for (int ni = 0; ni < 2; ++ni) {
            int n = colT + wn * 32 + ni * 16 + lm;
            #pragma unroll
            for (int i = 0; i < 8; ++i) {
                int m = rowT + wm * 64 + mi * 16 + lh * 8 + i;
                Q[(size_t)m * F_DIM + n] = (bf16_t)acc[mi][ni][i];
            }
        }
}

// ------- fused attention: scores = Q@K^T -> softmax -> out = A@V -------
__global__ __launch_bounds__(256) void k_attn(const bf16_t* __restrict__ Q,
                                              const bf16_t* __restrict__ K,
                                              const bf16_t* __restrict__ VT,
                                              float* __restrict__ out) {
    __shared__ bf16_t aTile[8][16][LDS_STRIDE];
    const int lane = threadIdx.x & 31;
    const int w    = threadIdx.x >> 5;      // wave owns 16 rows
    const int lm   = lane & 15;
    const int lh   = lane >> 4;
    const int rowBase = blockIdx.x * 128 + w * 16;

    // ---- Phase 1: scores (16 x 64) via 4 WMMA n-tiles, K=1024 ----
    v8f s[4];
    #pragma unroll
    for (int t = 0; t < 4; ++t) s[t] = (v8f){0.f,0.f,0.f,0.f,0.f,0.f,0.f,0.f};

    for (int kk = 0; kk < F_DIM; kk += 32) {
        Frag a;
        const bf16_t* p = Q + (size_t)(rowBase + lm) * F_DIM + kk + lh * 8;
        a.h[0] = *(const v8bf*)p;
        a.h[1] = *(const v8bf*)(p + 16);
        #pragma unroll
        for (int t = 0; t < 4; ++t) {
            Frag b;
            const bf16_t* q = K + (size_t)(t * 16 + lm) * F_DIM + kk + lh * 8;
            b.h[0] = *(const v8bf*)q;
            b.h[1] = *(const v8bf*)(q + 16);
            s[t] = __builtin_amdgcn_wmma_f32_16x16x32_bf16(
                false, a.v, false, b.v, (short)0, s[t], false, false);
        }
    }

    // ---- softmax over C=64: row m lives in one 16-lane group (c = t*16 + lane%16) ----
    #pragma unroll
    for (int i = 0; i < 8; ++i) {
        float mx = s[0][i];
        #pragma unroll
        for (int t = 1; t < 4; ++t) mx = fmaxf(mx, s[t][i]);
        #pragma unroll
        for (int d = 1; d < 16; d <<= 1) mx = fmaxf(mx, __shfl_xor(mx, d, 32));
        float p[4]; float sm = 0.f;
        #pragma unroll
        for (int t = 0; t < 4; ++t) { p[t] = __expf(s[t][i] - mx); sm += p[t]; }
        #pragma unroll
        for (int d = 1; d < 16; d <<= 1) sm += __shfl_xor(sm, d, 32);
        float inv = 1.f / sm;
        int r = lh * 8 + i;
        #pragma unroll
        for (int t = 0; t < 4; ++t)
            aTile[w][r][t * 16 + lm] = (bf16_t)(p[t] * inv);
    }
    __syncthreads();

    // ---- Phase 2: out = A @ V  (K=64 -> 2 WMMA per n-tile, 64 n-tiles) ----
    Frag af[2];
    const bf16_t* arow = &aTile[w][lm][0];
    af[0].h[0] = *(const v8bf*)(arow + lh * 8);
    af[0].h[1] = *(const v8bf*)(arow + lh * 8 + 16);
    af[1].h[0] = *(const v8bf*)(arow + 32 + lh * 8);
    af[1].h[1] = *(const v8bf*)(arow + 32 + lh * 8 + 16);

    for (int nt = 0; nt < 64; ++nt) {
        const bf16_t* vrow = VT + (size_t)(nt * 16 + lm) * C_DIM;
        Frag b0, b1;
        b0.h[0] = *(const v8bf*)(vrow + lh * 8);
        b0.h[1] = *(const v8bf*)(vrow + lh * 8 + 16);
        b1.h[0] = *(const v8bf*)(vrow + 32 + lh * 8);
        b1.h[1] = *(const v8bf*)(vrow + 32 + lh * 8 + 16);
        v8f acc = (v8f){0.f,0.f,0.f,0.f,0.f,0.f,0.f,0.f};
        acc = __builtin_amdgcn_wmma_f32_16x16x32_bf16(false, af[0].v, false, b0.v, (short)0, acc, false, false);
        acc = __builtin_amdgcn_wmma_f32_16x16x32_bf16(false, af[1].v, false, b1.v, (short)0, acc, false, false);
        #pragma unroll
        for (int i = 0; i < 8; ++i)
            out[(size_t)(rowBase + lh * 8 + i) * F_DIM + nt * 16 + lm] = acc[i];
    }
}

// ---------------------------------------------------------------------------
extern "C" void kernel_launch(void* const* d_in, const int* in_sizes, int n_in,
                              void* d_out, int out_size, void* d_ws, size_t ws_size,
                              hipStream_t stream) {
    const float* inputs = (const float*)d_in[0];   // (8,2048,1024) f32
    const float* WQ     = (const float*)d_in[1];   // (1024,1024)   f32
    const float* lab    = (const float*)d_in[2];   // (64,1024)     f32
    const float* WK     = (const float*)d_in[3];   // (1024,1024)   f32
    const float* WV     = (const float*)d_in[4];   // (1024,1024)   f32
    float* out = (float*)d_out;                    // (8,2048,1024) f32

    // workspace layout (bf16 elements)
    bf16_t* ws  = (bf16_t*)d_ws;
    bf16_t* Abf = ws;                              // 16384*1024
    bf16_t* WQT = ws + 16777216u;                  // 1024*1024
    bf16_t* Qbf = ws + 17825792u;                  // 16384*1024
    bf16_t* Kbf = ws + 34603008u;                  // 64*1024
    bf16_t* VTb = ws + 34668544u;                  // 1024*64

    k_convert_bf16<<<16384, 256, 0, stream>>>(inputs, Abf, (M_TOTAL * F_DIM) / 4);
    k_transpose_wq<<<dim3(32, 32), dim3(32, 8), 0, stream>>>(WQ, WQT);
    k_prep_kv<<<dim3(4, 64), 256, 0, stream>>>(lab, WK, WV, Kbf, VTb);
    k_gemm_q<<<dim3(M_TOTAL / 128, F_DIM / 128), 256, 0, stream>>>(Abf, WQT, Qbf);
    k_attn<<<M_TOTAL / 128, 256, 0, stream>>>(Qbf, Kbf, VTb, out);
}
